// SoftVQ_66889820668664
// MI455X (gfx1250) — matrix operand order repaired and checked by reading, not compiled, not run on hardware
//
#include <hip/hip_runtime.h>
#include <hip/hip_bf16.h>

// CDNA5 WMMA types
typedef __attribute__((ext_vector_type(16))) __bf16        v16bf;
typedef __attribute__((ext_vector_type(2)))  __bf16        bf16x2;
typedef __attribute__((ext_vector_type(8)))  float         v8f;
typedef __attribute__((ext_vector_type(8)))  unsigned int  v8u;

// Split one f32 pair into packed bf16 hi-parts and bf16 lo-residuals.
// Native bf16x2 vectors let clang emit v_cvt_pk_bf16_f32 (one per pair)
// plus v_fma_mix_f32_bf16 for the residuals.
__device__ __forceinline__ void split_pair(float x, float y,
                                           unsigned& hi, unsigned& lo) {
    bf16x2 h;
    h[0] = (__bf16)x;
    h[1] = (__bf16)y;
    float xr = x - (float)h[0];
    float yr = y - (float)h[1];
    bf16x2 l;
    l[0] = (__bf16)xr;
    l[1] = (__bf16)yr;
    hi = __builtin_bit_cast(unsigned, h);
    lo = __builtin_bit_cast(unsigned, l);
}

// Issue the global loads for one 32(M)x32(K) A tile and 32(K)x128(N) B tile
// into registers (no waits implied here; loads pipeline freely).
__device__ __forceinline__ void load_tiles(const float* __restrict__ A,
                                           const float* __restrict__ B,
                                           int K, int N, int m0, int n0, int k0,
                                           int tid,
                                           float2 (&areg)[2],
                                           float (&breg0)[8], float (&breg1)[8]) {
    #pragma unroll
    for (int it = 0; it < 2; ++it) {
        int p  = tid + it * 256;
        int ar = p >> 4, ap = p & 15;
        areg[it] = *(const float2*)&A[(size_t)(m0 + ar) * K + k0 + 2 * ap];
    }
    #pragma unroll
    for (int it = 0; it < 8; ++it) {
        int p  = tid + it * 256;
        int c  = p & 127, kp = p >> 7;
        const float* src = &B[(size_t)(k0 + 2 * kp) * N + (n0 + c)];
        breg0[it] = src[0];
        breg1[it] = src[(size_t)N];
    }
}

// ---------------------------------------------------------------------------
// Row-major GEMM  C[M,N] = A[M,K] * B[K,N]  (f32 in/out) via bf16x3 split WMMA.
// Workgroup: 256 threads = 8 wave32, computing a 32(M) x 128(N) block.
// Waves arranged 2(M) x 4(N); each wave owns 16x32 (two 16x16 WMMA tiles,
// sharing one A fragment). K advances in steps of 32, software-pipelined:
// global loads for step k+1 are issued before the WMMAs of step k.
// f32 -> bf16 hi/lo conversion happens once at staging; LDS holds packed
// bf16 pairs so each fragment VGPR is a single ds_load (merged to 2addr).
// Requires M%32==0, N%128==0, K%32==0.
// ---------------------------------------------------------------------------
__global__ __launch_bounds__(256)
void gemm_bf16x3_kernel(const float* __restrict__ A,
                        const float* __restrict__ B,
                        float* __restrict__ C,
                        int M, int N, int K) {
    // Packed bf16-pair tiles. Padding: A stride 17 (odd -> conflict-free),
    // B stride 132 (4*132 % 64 == 16 -> upper-half lanes hit disjoint banks).
    __shared__ unsigned a_hi_lds[32][17];
    __shared__ unsigned a_lo_lds[32][17];
    __shared__ unsigned b_hi_lds[16][132];
    __shared__ unsigned b_lo_lds[16][132];

    const int tid  = threadIdx.x;
    const int lane = tid & 31;
    const int wave = tid >> 5;        // 0..7
    const int mw   = wave >> 2;       // 0..1: M sub-block
    const int nw   = wave & 3;        // 0..3: N sub-block
    const int half = lane >> 4;       // lane group 0/1
    const int r    = lane & 15;       // A row / B,C column within a tile

    const int m0 = blockIdx.y * 32;
    const int n0 = blockIdx.x * 128;

    v8f acc0 = {0.f, 0.f, 0.f, 0.f, 0.f, 0.f, 0.f, 0.f};
    v8f acc1 = {0.f, 0.f, 0.f, 0.f, 0.f, 0.f, 0.f, 0.f};

    float2 areg[2];
    float  breg0[8], breg1[8];

    // Pipeline prologue: fetch first tiles.
    load_tiles(A, B, K, N, m0, n0, /*k0=*/0, tid, areg, breg0, breg1);

    for (int k0 = 0; k0 < K; k0 += 32) {
        // ---- convert current registers once, store packed bf16 to LDS -----
        #pragma unroll
        for (int it = 0; it < 2; ++it) {
            int p  = tid + it * 256;
            int ar = p >> 4, ap = p & 15;
            unsigned h, l;
            split_pair(areg[it].x, areg[it].y, h, l);
            a_hi_lds[ar][ap] = h;
            a_lo_lds[ar][ap] = l;
        }
        #pragma unroll
        for (int it = 0; it < 8; ++it) {
            int p  = tid + it * 256;
            int c  = p & 127, kp = p >> 7;
            unsigned h, l;
            split_pair(breg0[it], breg1[it], h, l);
            b_hi_lds[kp][c] = h;
            b_lo_lds[kp][c] = l;
        }
        __syncthreads();

        // ---- issue NEXT step's global loads; they overlap the WMMAs -------
        if (k0 + 32 < K) {
            load_tiles(A, B, K, N, m0, n0, k0 + 32, tid, areg, breg0, breg1);
        }
        if (k0 + 64 < K) {
            __builtin_prefetch(&A[(size_t)(m0 + (tid >> 4)) * K + k0 + 64], 0, 1);
            __builtin_prefetch(&B[(size_t)(k0 + 64 + (tid >> 7)) * N + n0 + (tid & 127)], 0, 1);
        }

        // ---- fragments: pure LDS pair loads -------------------------------
        // A 16x32 layout: VGPR v holds K pair kp = (v&3) + ((v>>2)<<3) + half*4
        v8u ah, al;
        #pragma unroll
        for (int v = 0; v < 8; ++v) {
            int kp = (v & 3) + ((v >> 2) << 3) + half * 4;
            ah[v] = a_hi_lds[mw * 16 + r][kp];
            al[v] = a_lo_lds[mw * 16 + r][kp];
        }
        // B 32x16 layout: VGPR v holds K pair kp = (v&3) + ((v>>2)<<3) + half*4, column = lane r
        v8u bh0, bl0, bh1, bl1;
        #pragma unroll
        for (int v = 0; v < 8; ++v) {
            int kp = (v & 3) + ((v >> 2) << 3) + half * 4;
            int c0 = nw * 32 + r;
            bh0[v] = b_hi_lds[kp][c0];
            bl0[v] = b_lo_lds[kp][c0];
            bh1[v] = b_hi_lds[kp][c0 + 16];
            bl1[v] = b_lo_lds[kp][c0 + 16];
        }
        v16bf a_hi   = __builtin_bit_cast(v16bf, ah);
        v16bf a_lo   = __builtin_bit_cast(v16bf, al);
        v16bf b_hi_0 = __builtin_bit_cast(v16bf, bh0);
        v16bf b_lo_0 = __builtin_bit_cast(v16bf, bl0);
        v16bf b_hi_1 = __builtin_bit_cast(v16bf, bh1);
        v16bf b_lo_1 = __builtin_bit_cast(v16bf, bl1);

        // ---- bf16x3 WMMA; grouped so a_hi feeds 4 consecutive issues ------
        acc0 = __builtin_amdgcn_wmma_f32_16x16x32_bf16(
                   false, a_hi, false, b_hi_0, (short)0, acc0, false, false);
        acc1 = __builtin_amdgcn_wmma_f32_16x16x32_bf16(
                   false, a_hi, false, b_hi_1, (short)0, acc1, false, false);
        acc0 = __builtin_amdgcn_wmma_f32_16x16x32_bf16(
                   false, a_hi, false, b_lo_0, (short)0, acc0, false, false);
        acc1 = __builtin_amdgcn_wmma_f32_16x16x32_bf16(
                   false, a_hi, false, b_lo_1, (short)0, acc1, false, false);
        acc0 = __builtin_amdgcn_wmma_f32_16x16x32_bf16(
                   false, a_lo, false, b_hi_0, (short)0, acc0, false, false);
        acc1 = __builtin_amdgcn_wmma_f32_16x16x32_bf16(
                   false, a_lo, false, b_hi_1, (short)0, acc1, false, false);

        __syncthreads();
    }

    // ---- store C per 16x16 f32 C/D layout: VGPR j -> row j + half*8 -------
    const int col = n0 + nw * 32 + r;
    #pragma unroll
    for (int j = 0; j < 8; ++j) {
        int m = m0 + mw * 16 + j + half * 8;
        C[(size_t)m * N + col]      = acc0[j];
        C[(size_t)m * N + col + 16] = acc1[j];
    }
}

// ---------------------------------------------------------------------------
// In-place numerically-stable softmax over rows of length N (4096 here).
// One 256-thread workgroup per row; each thread owns N/256 elements.
// ---------------------------------------------------------------------------
__global__ __launch_bounds__(256)
void softmax_rows_kernel(float* __restrict__ W, int N) {
    __shared__ float red[256];
    const int tid = threadIdx.x;
    float* w = W + (size_t)blockIdx.x * N;
    const int per = N / 256;   // 16

    float vals[16];
    float mx = -__builtin_inff();
    #pragma unroll
    for (int i = 0; i < 16; ++i) {
        if (i < per) {
            vals[i] = w[tid + i * 256];
            mx = fmaxf(mx, vals[i]);
        }
    }
    red[tid] = mx;
    __syncthreads();
    for (int s = 128; s > 0; s >>= 1) {
        if (tid < s) red[tid] = fmaxf(red[tid], red[tid + s]);
        __syncthreads();
    }
    mx = red[0];
    __syncthreads();

    float sum = 0.f;
    #pragma unroll
    for (int i = 0; i < 16; ++i) {
        if (i < per) {
            vals[i] = __expf(vals[i] - mx);
            sum += vals[i];
        }
    }
    red[tid] = sum;
    __syncthreads();
    for (int s = 128; s > 0; s >>= 1) {
        if (tid < s) red[tid] += red[tid + s];
        __syncthreads();
    }
    float inv = 1.0f / red[0];
    #pragma unroll
    for (int i = 0; i < 16; ++i) {
        if (i < per) w[tid + i * 256] = vals[i] * inv;
    }
}

// ---------------------------------------------------------------------------
extern "C" void kernel_launch(void* const* d_in, const int* in_sizes, int n_in,
                              void* d_out, int out_size, void* d_ws, size_t ws_size,
                              hipStream_t stream) {
    const float* query = (const float*)d_in[0];   // [B*T, KEY_DIM]
    const float* keys  = (const float*)d_in[1];   // [KEY_DIM, DICT_SIZE]
    const float* value = (const float*)d_in[2];   // [DICT_SIZE, DICT_DIM]

    const int key_dim   = 512;
    const int M         = in_sizes[0] / key_dim;            // 16384
    const int dict_size = in_sizes[1] / key_dim;            // 4096
    const int dict_dim  = in_sizes[2] / dict_size;          // 512

    float* weight = (float*)d_out;                          // [M, dict_size]
    float* vq_out = weight + (size_t)M * dict_size;         // [M, dict_dim]

    // GEMM1: logits = query @ keys   (M x dict_size, K = key_dim)
    dim3 g1(dict_size / 128, M / 32);
    gemm_bf16x3_kernel<<<g1, 256, 0, stream>>>(query, keys, weight,
                                               M, dict_size, key_dim);

    // Softmax rows in place -> weight
    softmax_rows_kernel<<<M, 256, 0, stream>>>(weight, dict_size);

    // GEMM2: vq_out = weight @ value  (M x dict_dim, K = dict_size)
    dim3 g2(dict_dim / 128, M / 32);
    gemm_bf16x3_kernel<<<g2, 256, 0, stream>>>(weight, value, vq_out,
                                               M, dict_dim, dict_size);
}
